// Attention_17334488007435
// MI455X (gfx1250) — compile-verified, matrix operand order checked by reading, not run on hardware
//
#include <hip/hip_runtime.h>
#include <hip/hip_bf16.h>
#include <math.h>

// ---------------------------------------------------------------------------
// Attention block for MI455X (gfx1250, wave32, WMMA).
// All GEMMs + attention use v_wmma_f32_16x16x32_bf16 (fp32 accumulate).
// Flash-attention streaming; K/V tiles staged cooperatively into LDS with
// global_load_async_to_lds_b128 (ASYNCcnt double-buffered).
// ---------------------------------------------------------------------------

typedef __bf16 bf16;
typedef __attribute__((ext_vector_type(16))) __bf16 v16bf;
typedef __attribute__((ext_vector_type(8)))  __bf16 v8bf;
typedef __attribute__((ext_vector_type(8)))  float  v8f;

#define WMMA_BF16(A, B, C) \
  __builtin_amdgcn_wmma_f32_16x16x32_bf16(false, (A), false, (B), (short)0, (C), false, false)

__device__ __forceinline__ v16bf ld16g(const bf16* p) {
  return *reinterpret_cast<const v16bf*>(p);
}

// butterfly reductions inside each 16-lane half (row stats of a C fragment)
__device__ __forceinline__ float rmax16(float v) {
#pragma unroll
  for (int m = 1; m <= 8; m <<= 1) v = fmaxf(v, __shfl_xor(v, m, 32));
  return v;
}
__device__ __forceinline__ float rsum16(float v) {
#pragma unroll
  for (int m = 1; m <= 8; m <<= 1) v += __shfl_xor(v, m, 32);
  return v;
}

// ---------------- constants -------------------------------------------------
#define NB    4
#define NC    256
#define NN    4096
#define NH    4
#define DH    32
#define HID   128
#define QKVO  384
#define SCALE_LOG2E (10.0f * 1.4426950408889634f)

// ---------------- K0a: weights fp32 -> bf16 ---------------------------------
__global__ void k0_weights(const float* __restrict__ wq, const float* __restrict__ wo,
                           bf16* __restrict__ wqb, bf16* __restrict__ wob) {
  int i = blockIdx.x * 256 + threadIdx.x;
  if (i < QKVO * NC) wqb[i] = (bf16)wq[i];
  if (i < NC * HID)  wob[i] = (bf16)wo[i];
}

// ---------------- K0b: x (b,c,n) fp32 -> x_t (b,n,c) bf16 (LDS tiled) -------
__global__ __launch_bounds__(1024) void k0_xt(const float* __restrict__ x,
                                              bf16* __restrict__ xt) {
  __shared__ bf16 tile[32][33];
  int b  = blockIdx.z;
  int c0 = blockIdx.y * 32, n0 = blockIdx.x * 32;
  int tx = threadIdx.x, ty = threadIdx.y;
  tile[ty][tx] = (bf16)x[((size_t)(b * NC + c0 + ty)) * NN + n0 + tx];
  __syncthreads();
  xt[((size_t)(b * NN + n0 + ty)) * NC + c0 + tx] = tile[tx][ty];
}

// ---------------- K1: QKV GEMM, wave = 16(o) x 64(n) tile -------------------
__global__ __launch_bounds__(256) void k1_qkv(const bf16* __restrict__ wbf,
                                              const bf16* __restrict__ xt,
                                              bf16* __restrict__ qt,
                                              bf16* __restrict__ kt,
                                              bf16* __restrict__ vdn) {
  int wid = threadIdx.x >> 5, lane = threadIdx.x & 31;
  int w   = blockIdx.x * 8 + wid;           // 4*24*64 = 6144 waves
  int nt  = w & 63;
  int tmp = w >> 6;
  int mt  = tmp % 24, b = tmp / 24;
  int o0 = mt * 16, n0 = nt * 64;
  int t = lane & 15, g = lane >> 4;

  v8f acc[4] = {};
  for (int k0 = 0; k0 < NC; k0 += 32) {
    v16bf a = ld16g(wbf + (size_t)(o0 + t) * NC + k0 + 16 * g);
#pragma unroll
    for (int f = 0; f < 4; ++f) {
      v16bf bx = ld16g(xt + ((size_t)(b * NN + n0 + 16 * f + t)) * NC + k0 + 16 * g);
      acc[f] = WMMA_BF16(a, bx, acc[f]);
    }
  }

  int s   = o0 >> 7;          // 0=q 1=k 2=v
  int rem = o0 & 127;
  int h = rem >> 5, d0 = rem & 31;   // d0 in {0,16}
  int bh = b * NH + h;
  if (s < 2) {
    bf16* dst = (s == 0) ? qt : kt;
#pragma unroll
    for (int f = 0; f < 4; ++f) {
      v8bf pk;
#pragma unroll
      for (int r = 0; r < 8; ++r) pk[r] = (bf16)acc[f][r];
      int n = n0 + 16 * f + t;
      *reinterpret_cast<v8bf*>(dst + ((size_t)(bh * NN + n)) * DH + d0 + 8 * g) = pk;
    }
  } else {
#pragma unroll
    for (int f = 0; f < 4; ++f) {
      int n = n0 + 16 * f + t;
#pragma unroll
      for (int r = 0; r < 8; ++r)
        vdn[((size_t)(bh * DH + d0 + r + 8 * g)) * NN + n] = (bf16)acc[f][r];
    }
  }
}

// ---------------- K2: l2-norm over n per (s,bh,d); rescale in place ---------
__global__ __launch_bounds__(256) void k2_norm(bf16* __restrict__ qt,
                                               bf16* __restrict__ kt) {
  __shared__ float red[256];
  __shared__ float scl[32];
  int s  = blockIdx.x >> 4;       // 0=q, 1=k
  int bh = blockIdx.x & 15;
  bf16* p = ((s == 0) ? qt : kt) + (size_t)bh * NN * DH;
  int d = threadIdx.x & 31, part = threadIdx.x >> 5;

  float ss = 0.f;
  for (int n = part; n < NN; n += 8) {
    float v = (float)p[(size_t)n * DH + d];
    ss += v * v;
  }
  red[threadIdx.x] = ss;
  __syncthreads();
  if (part == 0) {
    float tot = 0.f;
#pragma unroll
    for (int q = 0; q < 8; ++q) tot += red[q * 32 + d];
    float sc = 1.f / fmaxf(sqrtf(tot), 1e-12f);
    if (s == 0) sc *= SCALE_LOG2E;
    scl[d] = sc;
  }
  __syncthreads();
  float sc = scl[d];
  for (int n = part; n < NN; n += 8) {
    size_t idx = (size_t)n * DH + d;
    p[idx] = (bf16)((float)p[idx] * sc);
  }
}

// ---------------- K3: flash attention, wave = 16 queries --------------------
// Block = 8 waves sharing one (b,h). Per 32-key chunk each wave issues ONE
// global_load_async_to_lds_b128 (512B) to cooperatively stage K (2KB) + V
// (2KB) tiles; ASYNCcnt double-buffered; fragments read from LDS.
__global__ __launch_bounds__(256) void k3_attn(const bf16* __restrict__ qt,
                                               const bf16* __restrict__ kt,
                                               const bf16* __restrict__ vdn,
                                               bf16* __restrict__ outh) {
  // bytes [0,8192): per-wave P-transpose bufs; [8192,16384): KV double buffer
  __shared__ bf16 smem[8192] __attribute__((aligned(128)));
  int wid = threadIdx.x >> 5, lane = threadIdx.x & 31;
  int w  = blockIdx.x * 8 + wid;     // 16*256 = 4096 waves
  int bh = w >> 8;
  int i0 = (w & 255) << 4;
  int b = bh >> 2, h = bh & 3;
  int t = lane & 15, g = lane >> 4;
  bf16* lb = smem + wid * 512;       // elements; P-transpose buffer

  // LDS byte address of smem (flat-address low 32 bits == LDS byte offset)
  unsigned smem_lds = (unsigned)(size_t)(&smem[0]);

  // staging role: waves 0-3 -> K tile (contiguous 2KB), waves 4-7 -> V tile
  unsigned idx   = (unsigned)((wid & 3) * 32 + lane);   // 0..127 (16B pieces)
  bool     isV   = wid >= 4;
  unsigned piece = isV ? (2048u + (idx >> 2) * 64u + (idx & 3) * 16u)  // d row, seg
                       : (idx * 16u);
  const bf16* gK = kt  + (size_t)(bh * NN) * DH;  // + j0*32 elems for chunk
  const bf16* gV = vdn + (size_t)(bh * DH) * NN;  // row d: + d*4096 + j0
  unsigned gvoff_base = isV ? ((idx >> 2) * (unsigned)(NN * 2) + (idx & 3) * 16u)
                            : (idx * 16u);
  const bf16* gbase = isV ? gV : gK;

  auto stage = [&](int j0, int buf) {
    unsigned dst  = smem_lds + 8192u + (unsigned)buf * 4096u + piece;
    unsigned goff = gvoff_base + (unsigned)(isV ? (j0 * 2) : (j0 * 64));
    asm volatile("global_load_async_to_lds_b128 %0, %1, %2"
                 :: "v"(dst), "v"(goff), "s"(gbase) : "memory");
  };

  // Q A-fragment: lane = query row i0+t, K-half g, 16 contiguous bf16
  v16bf aq = ld16g(qt + ((size_t)(bh * NN + i0 + t)) * DH + 16 * g);

  float m[8], l[8];
  v8f o0 = {}, o1 = {};
#pragma unroll
  for (int r = 0; r < 8; ++r) { m[r] = -INFINITY; l[r] = 0.f; }

  stage(0, 0);
  for (int c = 0; c < 128; ++c) {
    int cn = (c + 1 < 128) ? (c + 1) : 127;   // last prefetch redundant (safe)
    stage(cn * 32, (c + 1) & 1);
    asm volatile("s_wait_asynccnt 1" ::: "memory");  // oldest (chunk c) done
    __syncthreads();                                  // all pieces visible

    const bf16* kb = smem + 4096 + (c & 1) * 2048;    // element offsets
    const bf16* vb = kb + 1024;
    v16bf bk0 = *reinterpret_cast<const v16bf*>(kb + t * 32 + 16 * g);
    v16bf bk1 = *reinterpret_cast<const v16bf*>(kb + (16 + t) * 32 + 16 * g);
    v16bf bv0 = *reinterpret_cast<const v16bf*>(vb + t * 32 + 16 * g);
    v16bf bv1 = *reinterpret_cast<const v16bf*>(vb + (16 + t) * 32 + 16 * g);

    v8f s0 = {}, s1 = {};
    s0 = WMMA_BF16(aq, bk0, s0);
    s1 = WMMA_BF16(aq, bk1, s1);

    // online softmax (exp2 domain; SCALE*log2e folded into q)
    float p0[8], p1[8];
#pragma unroll
    for (int r = 0; r < 8; ++r) {
      float rm = rmax16(fmaxf(s0[r], s1[r]));
      float mn = fmaxf(m[r], rm);
      float cc = exp2f(m[r] - mn);
      m[r] = mn;
      p0[r] = exp2f(s0[r] - mn);
      p1[r] = exp2f(s1[r] - mn);
      l[r] = l[r] * cc + rsum16(p0[r] + p1[r]);
      o0[r] *= cc;
      o1[r] *= cc;
    }

    // transpose P: C-layout -> A-layout via per-wave LDS buffer
#pragma unroll
    for (int r = 0; r < 8; ++r) {
      lb[(r + 8 * g) * 32 + t]      = (bf16)p0[r];
      lb[(r + 8 * g) * 32 + 16 + t] = (bf16)p1[r];
    }
    asm volatile("s_wait_dscnt 0" ::: "memory");
    v16bf ap = *reinterpret_cast<const v16bf*>(lb + t * 32 + 16 * g);

    o0 = WMMA_BF16(ap, bv0, o0);
    o1 = WMMA_BF16(ap, bv1, o1);

    __syncthreads();   // reads of buf done before it is restaged
  }

  // normalize rows and write (b, n, hid) via LDS transpose (coalesced b128s)
#pragma unroll
  for (int r = 0; r < 8; ++r) {
    float inv = 1.f / l[r];
    lb[(r + 8 * g) * 32 + t]      = (bf16)(o0[r] * inv);
    lb[(r + 8 * g) * 32 + 16 + t] = (bf16)(o1[r] * inv);
  }
  asm volatile("s_wait_dscnt 0" ::: "memory");
  v16bf ov = *reinterpret_cast<const v16bf*>(lb + t * 32 + 16 * g);
  *reinterpret_cast<v16bf*>(outh + ((size_t)(b * NN + i0 + t)) * HID + h * DH + 16 * g) = ov;
}

// ---------------- K4: output projection + bias, wave = 16(o) x 64(n) --------
__global__ __launch_bounds__(256) void k4_out(const bf16* __restrict__ wobf,
                                              const bf16* __restrict__ outh,
                                              const float* __restrict__ bias,
                                              float* __restrict__ y) {
  __shared__ float lf[8 * 16 * 64] __attribute__((aligned(16)));
  int wid = threadIdx.x >> 5, lane = threadIdx.x & 31;
  int w  = blockIdx.x * 8 + wid;   // 4*16*64 = 4096 waves
  int nt = w & 63;
  int tmp = w >> 6;
  int mt = tmp & 15, b = tmp >> 4;
  int o0 = mt * 16, n0 = nt * 64;
  int t = lane & 15, g = lane >> 4;

  v8f acc[4] = {};
  for (int k0 = 0; k0 < HID; k0 += 32) {
    v16bf a = ld16g(wobf + (size_t)(o0 + t) * HID + k0 + 16 * g);
#pragma unroll
    for (int f = 0; f < 4; ++f) {
      v16bf bx = ld16g(outh + ((size_t)(b * NN + n0 + 16 * f + t)) * HID + k0 + 16 * g);
      acc[f] = WMMA_BF16(a, bx, acc[f]);
    }
  }

  float* lw = lf + wid * 1024;
#pragma unroll
  for (int f = 0; f < 4; ++f)
#pragma unroll
    for (int r = 0; r < 8; ++r)
      lw[(r + 8 * g) * 64 + 16 * f + t] = acc[f][r];
  asm volatile("s_wait_dscnt 0" ::: "memory");

  int o = o0 + t;
  float bo = bias[o];
  const float* src = lw + t * 64 + 32 * g;
  float* dst = y + ((size_t)(b * NC + o)) * NN + n0 + 32 * g;
#pragma unroll
  for (int q = 0; q < 8; ++q) {
    float4 v4 = *reinterpret_cast<const float4*>(src + 4 * q);
    v4.x += bo; v4.y += bo; v4.z += bo; v4.w += bo;
    *reinterpret_cast<float4*>(dst + 4 * q) = v4;
  }
}

// ---------------- launch ----------------------------------------------------
extern "C" void kernel_launch(void* const* d_in, const int* in_sizes, int n_in,
                              void* d_out, int out_size, void* d_ws, size_t ws_size,
                              hipStream_t stream) {
  (void)in_sizes; (void)n_in; (void)out_size; (void)ws_size;
  const float* x    = (const float*)d_in[0];
  const float* wqkv = (const float*)d_in[1];
  const float* wout = (const float*)d_in[2];
  const float* bout = (const float*)d_in[3];
  float* y = (float*)d_out;

  char* ws = (char*)d_ws;
  bf16* xt  = (bf16*)(ws);                    // 8,388,608
  bf16* wqb = (bf16*)(ws + 8388608);          //   196,608
  bf16* wob = (bf16*)(ws + 8585216);          //    65,536
  bf16* qt  = (bf16*)(ws + 8650752);          // 4,194,304
  bf16* kt  = (bf16*)(ws + 12845056);         // 4,194,304
  bf16* vdn = (bf16*)(ws + 17039360);         // 4,194,304
  bf16* oh  = (bf16*)(ws + 21233664);         // 4,194,304  (total ~25.4 MB)

  k0_weights<<<384, 256, 0, stream>>>(wqkv, wout, wqb, wob);
  k0_xt<<<dim3(128, 8, 4), dim3(32, 32), 0, stream>>>(x, xt);
  k1_qkv<<<768, 256, 0, stream>>>(wqb, xt, qt, kt, vdn);
  k2_norm<<<32, 256, 0, stream>>>(qt, kt);
  k3_attn<<<512, 256, 0, stream>>>(qt, kt, vdn, oh);
  k4_out<<<512, 256, 0, stream>>>(wob, oh, bout, y);
}
